// SparseBottleneckAttention_76854144794866
// MI455X (gfx1250) — matrix operand order repaired and checked by reading, not compiled, not run on hardware
//
#include <hip/hip_runtime.h>
#include <hip/hip_bf16.h>
#include <math.h>

typedef _Float16 v16h __attribute__((ext_vector_type(16)));
typedef float    v8f  __attribute__((ext_vector_type(8)));

#define NUM_HEADS 8
#define HEAD_DIM  64
#define EMBED     512
#define SEQ       1024
#define BATCH     32
#define NPOOL     64
#define TOPK      16
#define MROWS     (BATCH * SEQ)   // 32768

__device__ __forceinline__ v8f wmma16(v16h a, v16h b, v8f c) {
  // D = A(16x32 f16) * B(32x16 f16) + C(16x16 f32)
  return __builtin_amdgcn_wmma_f32_16x16x32_f16(
      /*neg_a=*/false, a, /*neg_b=*/false, b,
      /*c_mod=*/(short)0, c, /*reuse_a=*/false, /*reuse_b=*/false);
}

// ---------------------------------------------------------------------------
// WMMA GEMM:  C[M,N] = act(A[M,K] @ W[K,N] + bias)
// A: f16 row-major.  Wt: f16 *transposed* (N x K) so fragments are contiguous.
// Block tile 128x128, per-wave tile 32x64 (8 WMMAs / wave / K-step),
// double-buffered LDS staging + global prefetch of K-tile t+2.
// OUTMODE: 0 = store f32, 1 = store f16, 2 = f32 += (residual add)
// ACT:     0 = none, 1 = exact GELU
// ---------------------------------------------------------------------------
template<int OUTMODE, int ACT>
__global__ __launch_bounds__(256)
void gemm_kernel(const _Float16* __restrict__ A, const _Float16* __restrict__ Wt,
                 const float* __restrict__ bias, float* __restrict__ Cf32,
                 _Float16* __restrict__ Cf16, int M, int N, int K)
{
  __shared__ __align__(16) _Float16 As[2][128][40];  // 128 rows x 32 halves (+8 pad)
  __shared__ __align__(16) _Float16 Bs[2][128][40];  // 128 cols(N) x 32 halves(K)
  const int tid  = threadIdx.x;
  const int lane = tid & 31;
  const int w    = tid >> 5;
  const int wm   = w & 3;            // wave row group: 32 rows each
  const int wn   = w >> 2;           // wave col group: 64 cols each
  const int bm0  = blockIdx.x * 128;
  const int bn0  = blockIdx.y * 128;
  const int sr   = tid >> 1;         // staging row 0..127
  const int sc   = (tid & 1) * 16;   // staging half-chunk 0 / 16

  v8f acc[2][4] = {};

  const _Float16* Ag = A  + (size_t)(bm0 + sr) * K + sc;
  const _Float16* Bg = Wt + (size_t)(bn0 + sr) * K + sc;

  // Stage K-tile 0 into buffer 0
  *(int4*)&As[0][sr][sc]     = *(const int4*)(Ag);
  *(int4*)&As[0][sr][sc + 8] = *(const int4*)(Ag + 8);
  *(int4*)&Bs[0][sr][sc]     = *(const int4*)(Bg);
  *(int4*)&Bs[0][sr][sc + 8] = *(const int4*)(Bg + 8);
  __syncthreads();

  const int fm = lane & 15;          // fragment row/col within 16
  const int kh = (lane >> 4) * 16;   // A-fragment K group
  const int bk = (lane >> 4) * 16;   // B-fragment K group
  const int KT = K / 32;

  for (int kt = 0; kt < KT; ++kt) {
    const int cur = kt & 1, nxt = cur ^ 1;
    if (kt + 1 < KT) {
      // Stage next K-tile into the other buffer while WMMAs run on this one
      const _Float16* Ag2 = Ag + (kt + 1) * 32;
      const _Float16* Bg2 = Bg + (kt + 1) * 32;
      *(int4*)&As[nxt][sr][sc]     = *(const int4*)(Ag2);
      *(int4*)&As[nxt][sr][sc + 8] = *(const int4*)(Ag2 + 8);
      *(int4*)&Bs[nxt][sr][sc]     = *(const int4*)(Bg2);
      *(int4*)&Bs[nxt][sr][sc + 8] = *(const int4*)(Bg2 + 8);
      if (kt + 2 < KT) {
        __builtin_prefetch(Ag + (kt + 2) * 32, 0, 0);  // global_prefetch_b8
        __builtin_prefetch(Bg + (kt + 2) * 32, 0, 0);
      }
    }
    // A fragments (contiguous K order, k = 16*khalf + e): lane m=l%16, khalf=l>>4
    v16h af[2];
#pragma unroll
    for (int mt = 0; mt < 2; ++mt) {
      const int am = wm * 32 + mt * 16 + fm;
      ((int4*)&af[mt])[0] = *(const int4*)&As[cur][am][kh];
      ((int4*)&af[mt])[1] = *(const int4*)&As[cur][am][kh + 8];
    }
    // B fragments: lane n=l%16, kgroup=l>>4, elem e -> K = 16*kgroup + e
#pragma unroll
    for (int nt = 0; nt < 4; ++nt) {
      const int bn = wn * 64 + nt * 16 + fm;
      v16h bf;
      ((int4*)&bf)[0] = *(const int4*)&Bs[cur][bn][bk];
      ((int4*)&bf)[1] = *(const int4*)&Bs[cur][bn][bk + 8];
      acc[0][nt] = wmma16(af[0], bf, acc[0][nt]);
      acc[1][nt] = wmma16(af[1], bf, acc[1][nt]);
    }
    __syncthreads();
  }

  // C/D layout: lane n = l%16, VGPR r -> row = r + 8*(l>>4)
  const int cn = lane & 15;
  const int mh = (lane >> 4) * 8;
#pragma unroll
  for (int mt = 0; mt < 2; ++mt) {
#pragma unroll
    for (int nt = 0; nt < 4; ++nt) {
      const int col = bn0 + wn * 64 + nt * 16 + cn;
      const float bv = bias ? bias[col] : 0.0f;
#pragma unroll
      for (int r = 0; r < 8; ++r) {
        const int row = bm0 + wm * 32 + mt * 16 + mh + r;
        float x = acc[mt][nt][r] + bv;
        if (ACT == 1) x = 0.5f * x * (1.0f + erff(x * 0.70710678118654752f));
        const size_t o = (size_t)row * N + col;
        if      (OUTMODE == 0) Cf32[o] = x;
        else if (OUTMODE == 1) Cf16[o] = (_Float16)x;
        else                   Cf32[o] += x;
      }
    }
  }
}

// ---------------------------------------------------------------------------
// LayerNorm over rows of 512 (one wave per row), f32 in -> f16 out
// ---------------------------------------------------------------------------
__global__ __launch_bounds__(256)
void ln_kernel(const float* __restrict__ x, const float* __restrict__ g,
               const float* __restrict__ b, _Float16* __restrict__ out)
{
  const int row  = blockIdx.x * 8 + (threadIdx.x >> 5);
  const int lane = threadIdx.x & 31;
  const float* xr = x + (size_t)row * EMBED;
  float4 v[4];
  float s = 0.0f, ss = 0.0f;
#pragma unroll
  for (int i = 0; i < 4; ++i) {
    v[i] = ((const float4*)xr)[lane + 32 * i];
    s  += v[i].x + v[i].y + v[i].z + v[i].w;
    ss += v[i].x * v[i].x + v[i].y * v[i].y + v[i].z * v[i].z + v[i].w * v[i].w;
  }
#pragma unroll
  for (int m = 16; m >= 1; m >>= 1) {
    s  += __shfl_xor(s,  m, 32);
    ss += __shfl_xor(ss, m, 32);
  }
  const float mean = s * (1.0f / EMBED);
  const float var  = ss * (1.0f / EMBED) - mean * mean;
  const float rs   = rsqrtf(var + 1e-3f);
  _Float16* outr = out + (size_t)row * EMBED;
#pragma unroll
  for (int i = 0; i < 4; ++i) {
    const int c0 = (lane + 32 * i) * 4;
    outr[c0 + 0] = (_Float16)((v[i].x - mean) * rs * g[c0 + 0] + b[c0 + 0]);
    outr[c0 + 1] = (_Float16)((v[i].y - mean) * rs * g[c0 + 1] + b[c0 + 1]);
    outr[c0 + 2] = (_Float16)((v[i].z - mean) * rs * g[c0 + 2] + b[c0 + 2]);
    outr[c0 + 3] = (_Float16)((v[i].w - mean) * rs * g[c0 + 3] + b[c0 + 3]);
  }
}

// 4x4 average pool of K and V heads: (B,S,C) -> (B,H,64,64)
__global__ void pool_kernel(const _Float16* __restrict__ kk, const _Float16* __restrict__ vv,
                            _Float16* __restrict__ pk, _Float16* __restrict__ pv)
{
  const int d = threadIdx.x;      // 0..63
  const int p = blockIdx.x;       // 0..63
  const int h = blockIdx.y;
  const int b = blockIdx.z;
  const int ph = p >> 3, pw = p & 7;
  float sk = 0.0f, sv = 0.0f;
  for (int dy = 0; dy < 4; ++dy)
    for (int dx = 0; dx < 4; ++dx) {
      const int s = (ph * 4 + dy) * 32 + (pw * 4 + dx);
      const size_t o = ((size_t)b * SEQ + s) * EMBED + h * HEAD_DIM + d;
      sk += (float)kk[o]; sv += (float)vv[o];
    }
  const size_t po = (((size_t)b * NUM_HEADS + h) * NPOOL + p) * HEAD_DIM + d;
  pk[po] = (_Float16)(sk * 0.0625f);
  pv[po] = (_Float16)(sv * 0.0625f);
}

__global__ void init_red_kernel(unsigned* __restrict__ red) {
  const int i = blockIdx.x * 256 + threadIdx.x;
  if (i < BATCH * NPOOL) red[i] = 0u;   // smallest ordered-uint = identity for max
}

__device__ __forceinline__ unsigned ford(float f) {
  unsigned u = __float_as_uint(f);
  return (u & 0x80000000u) ? ~u : (u | 0x80000000u);   // monotone float ordering
}

// scores = q . pk^T (16 rows of s per wave), max over rows, atomic-max into red[b][p]
__global__ __launch_bounds__(32)
void scores_kernel(const _Float16* __restrict__ q, const _Float16* __restrict__ pk,
                   unsigned* __restrict__ red)
{
  const int st = blockIdx.x, h = blockIdx.y, b = blockIdx.z;
  const int lane = threadIdx.x;
  const int m  = lane & 15;
  const int kh = (lane >> 4) * 16;
  const _Float16* qrow = q + ((size_t)b * SEQ + st * 16 + m) * EMBED + h * HEAD_DIM;
  v16h a[2];
#pragma unroll
  for (int ks = 0; ks < 2; ++ks) {
    ((int4*)&a[ks])[0] = *(const int4*)(qrow + ks * 32 + kh);
    ((int4*)&a[ks])[1] = *(const int4*)(qrow + ks * 32 + kh + 8);
  }
  const int n  = lane & 15;
  const int bk = (lane >> 4) * 16;
  const _Float16* pkb = pk + ((size_t)b * NUM_HEADS + h) * NPOOL * HEAD_DIM;
  v8f acc[4] = {};
#pragma unroll
  for (int nt = 0; nt < 4; ++nt) {
#pragma unroll
    for (int ks = 0; ks < 2; ++ks) {
      // B[k=d][n=p]: column n is pk row (contiguous 32B)
      const _Float16* pr = pkb + (size_t)(nt * 16 + n) * HEAD_DIM + ks * 32 + bk;
      v16h bf;
      ((int4*)&bf)[0] = *(const int4*)pr;
      ((int4*)&bf)[1] = *(const int4*)(pr + 8);
      acc[nt] = wmma16(a[ks], bf, acc[nt]);
    }
  }
#pragma unroll
  for (int nt = 0; nt < 4; ++nt) {
    float mx = acc[nt][0];
#pragma unroll
    for (int r = 1; r < 8; ++r) mx = fmaxf(mx, acc[nt][r]);
    mx = fmaxf(mx, __shfl_xor(mx, 16, 32));         // combine row halves
    if (lane < 16) atomicMax(&red[b * NPOOL + nt * 16 + lane], ford(mx));
  }
}

// stable top-16 of 64 (ordered-uint compare preserves float order; ties -> lowest idx)
__global__ void topk_kernel(const unsigned* __restrict__ red, int* __restrict__ idx) {
  const int b = blockIdx.x;
  if (threadIdx.x != 0) return;
  unsigned v[NPOOL];
  for (int i = 0; i < NPOOL; ++i) v[i] = red[b * NPOOL + i];
  for (int j = 0; j < TOPK; ++j) {
    int arg = 0; unsigned best = v[0];
    for (int i = 1; i < NPOOL; ++i) if (v[i] > best) { best = v[i]; arg = i; }
    idx[b * TOPK + j] = arg;
    v[arg] = 0u;
  }
}

// gather sel_k (row-major) and sel_v (transposed: d-major) for the 2nd WMMA
__global__ void gather_kernel(const _Float16* __restrict__ pk, const _Float16* __restrict__ pv,
                              const int* __restrict__ idx, _Float16* __restrict__ selk,
                              _Float16* __restrict__ selvT)
{
  const int d = threadIdx.x;   // 0..63
  const int j = blockIdx.x;    // 0..15
  const int h = blockIdx.y;
  const int b = blockIdx.z;
  const int p = idx[b * TOPK + j];
  const size_t bh  = (size_t)b * NUM_HEADS + h;
  const size_t src = (bh * NPOOL + p) * HEAD_DIM + d;
  selk [(bh * TOPK + j) * HEAD_DIM + d]       = pk[src];
  selvT[ bh * HEAD_DIM * TOPK + d * TOPK + j] = pv[src];
}

// fused sparse attention tile: logits (WMMA) -> softmax -> attn @ sel_v (WMMA) -> += residual
__global__ __launch_bounds__(32)
void attn_kernel(const _Float16* __restrict__ q, const _Float16* __restrict__ selk,
                 const _Float16* __restrict__ selvT, float* __restrict__ tokens)
{
  __shared__ __align__(16) _Float16 P[16][16];
  const int st = blockIdx.x, h = blockIdx.y, b = blockIdx.z;
  const int lane = threadIdx.x;
  const int m    = lane & 15;
  const int half = lane >> 4;
  const int kh   = half * 8;
  const _Float16* qrow = q + ((size_t)b * SEQ + st * 16 + m) * EMBED + h * HEAD_DIM;
  v16h a[2];
#pragma unroll
  for (int ks = 0; ks < 2; ++ks) {
    ((int4*)&a[ks])[0] = *(const int4*)(qrow + ks * 32 + kh);
    ((int4*)&a[ks])[1] = *(const int4*)(qrow + ks * 32 + 16 + kh);
  }
  const size_t bh = (size_t)b * NUM_HEADS + h;
  const int n  = lane & 15;
  const int bk = half * 8;
  const _Float16* skb = selk + bh * TOPK * HEAD_DIM;
  v8f accL = {};
#pragma unroll
  for (int ks = 0; ks < 2; ++ks) {
    const _Float16* pr = skb + (size_t)n * HEAD_DIM + ks * 32 + bk;
    v16h bf;
    ((int4*)&bf)[0] = *(const int4*)pr;
    ((int4*)&bf)[1] = *(const int4*)(pr + 16);
    accL = wmma16(a[ks], bf, accL);
  }
  // row softmax: lane holds col n for rows half*8 + r ; reduce across 16-lane group
#pragma unroll
  for (int r = 0; r < 8; ++r) {
    float x = accL[r] * 0.125f;                    // HEAD_DIM^-0.5
    float mx = x;
#pragma unroll
    for (int mm = 1; mm < 16; mm <<= 1) mx = fmaxf(mx, __shfl_xor(mx, mm, 32));
    float e = expf(x - mx);
    float sum = e;
#pragma unroll
    for (int mm = 1; mm < 16; mm <<= 1) sum += __shfl_xor(sum, mm, 32);
    P[half * 8 + r][n] = (_Float16)(e / sum);
  }
  __syncthreads();
  // A fragment for attn @ sel_v: 16x32 with K=16..31 zero-padded
  const int4 z = {0, 0, 0, 0};
  v16h pa;
  ((int4*)&pa)[0] = half ? z : *(const int4*)&P[m][0];   // half 0: K = 0..7 ; half 1: zero
  ((int4*)&pa)[1] = half ? z : *(const int4*)&P[m][8];   // half 0: K = 8..15 ; half 1: zero
  const _Float16* svb = selvT + bh * HEAD_DIM * TOPK;
  v8f accO[4] = {};
#pragma unroll
  for (int nt = 0; nt < 4; ++nt) {
    v16h bf;
    if (half == 0) {                               // kgroup 0: K=0..15 real
      const _Float16* pr = svb + (size_t)(nt * 16 + n) * TOPK;
      ((int4*)&bf)[0] = *(const int4*)pr;
      ((int4*)&bf)[1] = *(const int4*)(pr + 8);
    } else {                                       // kgroup 1: zero-padded K
      ((int4*)&bf)[0] = z;
      ((int4*)&bf)[1] = z;
    }
    accO[nt] = wmma16(pa, bf, accO[nt]);
  }
#pragma unroll
  for (int nt = 0; nt < 4; ++nt) {
    const int d = nt * 16 + n;
#pragma unroll
    for (int r = 0; r < 8; ++r) {
      const int s = st * 16 + half * 8 + r;
      const size_t o = ((size_t)b * SEQ + s) * EMBED + h * HEAD_DIM + d;
      tokens[o] += accO[nt][r];                    // block owns these rows/cols
    }
  }
}

__global__ void f32tof16_kernel(const float* __restrict__ in, _Float16* __restrict__ out, int n) {
  const int i = blockIdx.x * 256 + threadIdx.x;
  if (i < n) out[i] = (_Float16)in[i];
}

// convert W[K,N] f32 -> Wt[N,K] f16 (transposed for contiguous B fragments)
__global__ void wconv_kernel(const float* __restrict__ W, _Float16* __restrict__ Wt, int K, int N) {
  const int i = blockIdx.x * 256 + threadIdx.x;
  if (i < K * N) {
    const int k = i / N, n = i % N;
    Wt[(size_t)n * K + k] = (_Float16)W[i];
  }
}

extern "C" void kernel_launch(void* const* d_in, const int* in_sizes, int n_in,
                              void* d_out, int out_size, void* d_ws, size_t ws_size,
                              hipStream_t stream)
{
  (void)in_sizes; (void)n_in; (void)out_size; (void)ws_size;
  const float* feat = (const float*)d_in[0];
  const float* Win  = (const float*)d_in[1];
  const float* b_in = (const float*)d_in[2];
  const float* Wq   = (const float*)d_in[3];
  const float* Wk   = (const float*)d_in[4];
  const float* Wv   = (const float*)d_in[5];
  const float* g1   = (const float*)d_in[6];
  const float* be1  = (const float*)d_in[7];
  const float* g2   = (const float*)d_in[8];
  const float* be2  = (const float*)d_in[9];
  const float* Wf1  = (const float*)d_in[10];
  const float* bf1  = (const float*)d_in[11];
  const float* Wf2  = (const float*)d_in[12];
  const float* bf2  = (const float*)d_in[13];
  const float* Wout = (const float*)d_in[14];
  const float* bout = (const float*)d_in[15];

  char* ws = (char*)d_ws;
  size_t off = 0;
  auto alloc = [&](size_t bytes) -> void* {
    void* p = ws + off;
    off = (off + bytes + 255) & ~(size_t)255;
    return p;
  };
  _Float16* wt_in  = (_Float16*)alloc((size_t)EMBED * EMBED * 2);
  _Float16* wt_q   = (_Float16*)alloc((size_t)EMBED * EMBED * 2);
  _Float16* wt_k   = (_Float16*)alloc((size_t)EMBED * EMBED * 2);
  _Float16* wt_v   = (_Float16*)alloc((size_t)EMBED * EMBED * 2);
  _Float16* wt_out = (_Float16*)alloc((size_t)EMBED * EMBED * 2);
  _Float16* wt_f1  = (_Float16*)alloc((size_t)EMBED * 2 * EMBED * 2);
  _Float16* wt_f2  = (_Float16*)alloc((size_t)2 * EMBED * EMBED * 2);
  float*    tokens = (float*)   alloc((size_t)MROWS * EMBED * 4);
  _Float16* t16    = (_Float16*)alloc((size_t)MROWS * EMBED * 2);
  _Float16* q16    = (_Float16*)alloc((size_t)MROWS * EMBED * 2);
  _Float16* k16    = (_Float16*)alloc((size_t)MROWS * EMBED * 2);
  _Float16* v16    = (_Float16*)alloc((size_t)MROWS * EMBED * 2);
  _Float16* pk     = (_Float16*)alloc((size_t)BATCH * NUM_HEADS * NPOOL * HEAD_DIM * 2);
  _Float16* pv     = (_Float16*)alloc((size_t)BATCH * NUM_HEADS * NPOOL * HEAD_DIM * 2);
  unsigned* red    = (unsigned*)alloc((size_t)BATCH * NPOOL * 4);
  int*      idx    = (int*)     alloc((size_t)BATCH * TOPK * 4);
  _Float16* selk   = (_Float16*)alloc((size_t)BATCH * NUM_HEADS * TOPK * HEAD_DIM * 2);
  _Float16* selvT  = (_Float16*)alloc((size_t)BATCH * NUM_HEADS * TOPK * HEAD_DIM * 2);
  _Float16* h16    = (_Float16*)alloc((size_t)MROWS * 2 * EMBED * 2);
  _Float16* x16    = q16;   // reuse q buffer for f16 feature map before QKV

  const int M = MROWS;
  // 1) weights -> f16 transposed
  wconv_kernel<<<(EMBED * EMBED) / 256, 256, 0, stream>>>(Win,  wt_in,  EMBED, EMBED);
  wconv_kernel<<<(EMBED * EMBED) / 256, 256, 0, stream>>>(Wq,   wt_q,   EMBED, EMBED);
  wconv_kernel<<<(EMBED * EMBED) / 256, 256, 0, stream>>>(Wk,   wt_k,   EMBED, EMBED);
  wconv_kernel<<<(EMBED * EMBED) / 256, 256, 0, stream>>>(Wv,   wt_v,   EMBED, EMBED);
  wconv_kernel<<<(EMBED * EMBED) / 256, 256, 0, stream>>>(Wout, wt_out, EMBED, EMBED);
  wconv_kernel<<<(EMBED * 2 * EMBED) / 256, 256, 0, stream>>>(Wf1, wt_f1, EMBED, 2 * EMBED);
  wconv_kernel<<<(EMBED * 2 * EMBED) / 256, 256, 0, stream>>>(Wf2, wt_f2, 2 * EMBED, EMBED);
  // 2) input projection
  f32tof16_kernel<<<(M * EMBED) / 256, 256, 0, stream>>>(feat, x16, M * EMBED);
  gemm_kernel<0, 0><<<dim3(M / 128, EMBED / 128), 256, 0, stream>>>(
      x16, wt_in, b_in, tokens, nullptr, M, EMBED, EMBED);
  // 3) LN1 + QKV
  ln_kernel<<<M / 8, 256, 0, stream>>>(tokens, g1, be1, t16);
  gemm_kernel<1, 0><<<dim3(M / 128, EMBED / 128), 256, 0, stream>>>(
      t16, wt_q, nullptr, nullptr, q16, M, EMBED, EMBED);
  gemm_kernel<1, 0><<<dim3(M / 128, EMBED / 128), 256, 0, stream>>>(
      t16, wt_k, nullptr, nullptr, k16, M, EMBED, EMBED);
  gemm_kernel<1, 0><<<dim3(M / 128, EMBED / 128), 256, 0, stream>>>(
      t16, wt_v, nullptr, nullptr, v16, M, EMBED, EMBED);
  // 4) pooled K/V, scores-max, top-k, gather
  pool_kernel<<<dim3(NPOOL, NUM_HEADS, BATCH), HEAD_DIM, 0, stream>>>(k16, v16, pk, pv);
  init_red_kernel<<<(BATCH * NPOOL + 255) / 256, 256, 0, stream>>>(red);
  scores_kernel<<<dim3(SEQ / 16, NUM_HEADS, BATCH), 32, 0, stream>>>(q16, pk, red);
  topk_kernel<<<BATCH, 32, 0, stream>>>(red, idx);
  gather_kernel<<<dim3(TOPK, NUM_HEADS, BATCH), HEAD_DIM, 0, stream>>>(pk, pv, idx, selk, selvT);
  // 5) sparse attention (+ residual accumulate into tokens)
  attn_kernel<<<dim3(SEQ / 16, NUM_HEADS, BATCH), 32, 0, stream>>>(q16, selk, selvT, tokens);
  // 6) LN2 + FFN (GELU) + residual
  ln_kernel<<<M / 8, 256, 0, stream>>>(tokens, g2, be2, t16);
  gemm_kernel<1, 1><<<dim3(M / 128, (2 * EMBED) / 128), 256, 0, stream>>>(
      t16, wt_f1, bf1, nullptr, h16, M, 2 * EMBED, EMBED);
  gemm_kernel<2, 0><<<dim3(M / 128, EMBED / 128), 256, 0, stream>>>(
      h16, wt_f2, bf2, tokens, nullptr, M, EMBED, 2 * EMBED);
  // 7) output projection
  f32tof16_kernel<<<(M * EMBED) / 256, 256, 0, stream>>>(tokens, t16, M * EMBED);
  gemm_kernel<0, 0><<<dim3(M / 128, EMBED / 128), 256, 0, stream>>>(
      t16, wt_out, bout, (float*)d_out, nullptr, M, EMBED, EMBED);
}
